// LSTMEncoder_77713138254160
// MI455X (gfx1250) — matrix operand order repaired and checked by reading, not compiled, not run on hardware
//
#include <hip/hip_runtime.h>

// LSTM encoder fused kernel for MI455X (gfx1250, wave32, WMMA).
// B=1024, T=2048, D=5, H=64, OUT=128.
// 64 workgroups x 128 threads (4 waves); each WG owns a 16-row batch tile and
// runs the full T=2048 recurrence for both layers.
//  - Layer-2 weight B-fragments (Wih1, Whh1): pinned in VGPRs (128 regs).
//  - Layer-1 recurrent weights (Whh0): f16-staged in LDS, fragments loaded
//    per step (latency hidden under the WMMA stream).
//  - x staging + layer-1 input weights kept as packed f16 (v_fma_mix_f32 for
//    the x-projection) to hold total allocation < 256 VGPRs (no MSB churn).

#define T_STEPS 2048
#define D_IN    5
#define XPITCH  8        // padded f16 row stride for x staging
#define H_DIM   64
#define G_DIM   256      // 4*H
#define OUT_DIM 128
#define TILE_B  16
#define NTHREADS 128

typedef _Float16 v16h __attribute__((ext_vector_type(16)));
typedef _Float16 v8h  __attribute__((ext_vector_type(8)));
typedef float    v8f  __attribute__((ext_vector_type(8)));

__device__ __forceinline__ v8f vzero8f() {
  v8f z = {0.f,0.f,0.f,0.f,0.f,0.f,0.f,0.f};
  return z;
}

// sigmoid(x) = 1/(1+2^(-x*log2(e)))
__device__ __forceinline__ float fast_sig(float x) {
  float e = __builtin_amdgcn_exp2f(-1.44269504f * x);
  return __builtin_amdgcn_rcpf(1.0f + e);
}
// tanh(x) = 2/(1+2^(-2x*log2(e))) - 1
__device__ __forceinline__ float fast_tanh(float x) {
  float e = __builtin_amdgcn_exp2f(-2.88539008f * x);
  return 2.0f * __builtin_amdgcn_rcpf(1.0f + e) - 1.0f;
}

// A-fragment (16x32 f16) from an LDS buffer laid out [16 rows][64 cols].
// A layout (contiguous, matches B fragments): lane half kh = lane>>4;
//   elements 0..15 -> K = kbase + kh*16 + e
__device__ __forceinline__ v16h load_a_frag(const _Float16* buf, int lane, int kbase) {
  int m  = lane & 15;
  int kh = lane >> 4;
  return *(const v16h*)(buf + m * H_DIM + kbase + kh * 16);
}

// B-fragment (32x16 f16) from LDS weights stored row-major [g][64] f16.
// B layout used here (contiguous, matches A): lanes 0-15 hold K=kbase..+15 for column N=lane;
// lanes 16-31 hold K=kbase+16..+31 -> one contiguous 32B run per lane.
__device__ __forceinline__ v16h load_b_frag_lds(const _Float16* w, int g0, int lane, int kbase) {
  int n  = lane & 15;
  int kh = lane >> 4;
  return *(const v16h*)(w + (g0 + n) * H_DIM + kbase + kh * 16);
}

// Same B-fragment built directly from GLOBAL f32 weights [g][64] row-major.
__device__ __forceinline__ v16h load_b_frag_g(const float* __restrict__ w,
                                              int g0, int lane, int kbase) {
  const int n  = lane & 15;
  const int kh = lane >> 4;
  const float* p = w + (size_t)(g0 + n) * H_DIM + kbase + kh * 16;
  v16h r;
  #pragma unroll
  for (int e = 0; e < 16; ++e) r[e] = (_Float16)p[e];
  return r;
}

__device__ __forceinline__ v8f wmma_f16(v16h a, v16h b, v8f c) {
  return __builtin_amdgcn_wmma_f32_16x16x32_f16(
      /*neg_a=*/false, a, /*neg_b=*/false, b,
      /*c_mod=*/(short)0, c, /*reuse_a=*/false, /*reuse_b=*/false);
}

__global__ __launch_bounds__(NTHREADS, 1)
void lstm_encoder_kernel(const float* __restrict__ x,
                         const float* __restrict__ Wih0, const float* __restrict__ Whh0,
                         const float* __restrict__ bih0, const float* __restrict__ bhh0,
                         const float* __restrict__ Wih1, const float* __restrict__ Whh1,
                         const float* __restrict__ bih1, const float* __restrict__ bhh1,
                         const float* __restrict__ Wfc,  const float* __restrict__ bfc,
                         float* __restrict__ out)
{
  // ---- LDS ----
  __shared__ __align__(32) _Float16 s_wh0[G_DIM * H_DIM];    // Whh0 [g][k] f16 (32KB)
  __shared__ __align__(32) _Float16 s_h1[2][TILE_B * H_DIM]; // double-buffered h1 (f16)
  __shared__ __align__(32) _Float16 s_h2[2][TILE_B * H_DIM]; // double-buffered h2 (f16)
  __shared__ __align__(32) _Float16 s_x[2][TILE_B * XPITCH]; // double-buffered x(t), f16

  const int tid  = threadIdx.x;
  const int lane = tid & 31;
  const int q    = tid >> 5;           // wave id 0..3 -> N-quarter
  const int b0g  = blockIdx.x * TILE_B;

  const int n  = lane & 15;
  const int hi = lane >> 4;
  const int u  = q * 16 + n;           // hidden unit column 0..63 owned by this lane

  // ---- stage Whh0 -> LDS (f16) ; init LDS state ----
  for (int i = tid; i < G_DIM * H_DIM; i += NTHREADS) s_wh0[i] = (_Float16)Whh0[i];
  for (int i = tid; i < TILE_B * H_DIM; i += NTHREADS) {
    s_h1[0][i] = (_Float16)0.f;
    s_h2[0][i] = (_Float16)0.f;
  }
  for (int i = tid; i < TILE_B * D_IN; i += NTHREADS) {
    int b = i / D_IN, d = i % D_IN;
    s_x[0][b * XPITCH + d] = (_Float16)x[(size_t)(b0g + b) * (T_STEPS * D_IN) + d];
  }

  // ---- per-lane constants straight from global (one-time) ----
  float bi0 = bih0[0*H_DIM + u] + bhh0[0*H_DIM + u];
  float bf0 = bih0[1*H_DIM + u] + bhh0[1*H_DIM + u];
  float bg0 = bih0[2*H_DIM + u] + bhh0[2*H_DIM + u];
  float bo0 = bih0[3*H_DIM + u] + bhh0[3*H_DIM + u];
  float bi1 = bih1[0*H_DIM + u] + bhh1[0*H_DIM + u];
  float bf1 = bih1[1*H_DIM + u] + bhh1[1*H_DIM + u];
  float bg1 = bih1[2*H_DIM + u] + bhh1[2*H_DIM + u];
  float bo1 = bih1[3*H_DIM + u] + bhh1[3*H_DIM + u];
  _Float16 wI[D_IN], wF[D_IN], wG[D_IN], wO[D_IN];   // packed f16 (v_fma_mix inputs)
  #pragma unroll
  for (int d = 0; d < D_IN; ++d) {
    wI[d] = (_Float16)Wih0[(0*H_DIM + u) * D_IN + d];
    wF[d] = (_Float16)Wih0[(1*H_DIM + u) * D_IN + d];
    wG[d] = (_Float16)Wih0[(2*H_DIM + u) * D_IN + d];
    wO[d] = (_Float16)Wih0[(3*H_DIM + u) * D_IN + d];
  }

  // ---- layer-2 weight B-fragments pinned in VGPRs (16 x v16h = 128 VGPRs) ----
  v16h wbi1[4][2];   // layer-2 input     Wih1^T tiles  [grp][kstep]
  v16h wbh1[4][2];   // layer-2 recurrent Whh1^T tiles  [grp][kstep]
  #pragma unroll
  for (int grp = 0; grp < 4; ++grp) {
    const int g0 = (grp * 4 + q) * 16;   // gate tile base: col = grp*64 + u - n
    #pragma unroll
    for (int ks = 0; ks < 2; ++ks) {
      wbi1[grp][ks] = load_b_frag_g(Wih1, g0, lane, ks * 32);
      wbh1[grp][ks] = load_b_frag_g(Whh1, g0, lane, ks * 32);
    }
  }
  __syncthreads();

  float c1[8], c2[8];
  #pragma unroll
  for (int r = 0; r < 8; ++r) { c1[r] = 0.f; c2[r] = 0.f; }

  int p = 0;
  for (int t = 0; t < T_STEPS; ++t) {
    // ---------- phase 1: gates0 = h1_prev @ Whh0^T (ntiles grp*4+q) ----------
    v8f acc[4];
    #pragma unroll
    for (int g = 0; g < 4; ++g) acc[g] = vzero8f();
    {
      v16h a0 = load_a_frag(s_h1[p], lane, 0);
      v16h a1 = load_a_frag(s_h1[p], lane, 32);
      #pragma unroll
      for (int grp = 0; grp < 4; ++grp) {
        const int g0 = (grp * 4 + q) * 16;
        acc[grp] = wmma_f16(a0, load_b_frag_lds(s_wh0, g0, lane, 0),  acc[grp]);
        acc[grp] = wmma_f16(a1, load_b_frag_lds(s_wh0, g0, lane, 32), acc[grp]);
      }
    }

    // prefetch x(t+1) into the other x buffer (uniform branch; no WMMA inside)
    if (t + 1 < T_STEPS) {
      for (int i = tid; i < TILE_B * D_IN; i += NTHREADS) {
        int b = i / D_IN, d = i % D_IN;
        s_x[p ^ 1][b * XPITCH + d] =
            (_Float16)x[(size_t)(b0g + b) * (T_STEPS * D_IN) + (size_t)(t + 1) * D_IN + d];
      }
    }

    // ---------- phase 2: layer-1 elementwise (mix-FMA x-proj + activations + cell) ----------
    {
      _Float16 xr[8][D_IN];
      #pragma unroll
      for (int r = 0; r < 8; ++r) {
        const int b = hi * 8 + r;
        #pragma unroll
        for (int d = 0; d < D_IN; ++d) xr[r][d] = s_x[p][b * XPITCH + d];
      }
      #pragma unroll
      for (int r = 0; r < 8; ++r) {
        float xi = bi0, xf = bf0, xg = bg0, xo = bo0;
        #pragma unroll
        for (int d = 0; d < D_IN; ++d) {
          const float xv = (float)xr[r][d];
          xi += (float)wI[d] * xv; xf += (float)wF[d] * xv;
          xg += (float)wG[d] * xv; xo += (float)wO[d] * xv;
        }
        const float iv = fast_sig(acc[0][r] + xi);
        const float fv = fast_sig(acc[1][r] + xf);
        const float gv = fast_tanh(acc[2][r] + xg);
        const float ov = fast_sig(acc[3][r] + xo);
        const float c  = fv * c1[r] + iv * gv;
        c1[r] = c;
        const float h = ov * fast_tanh(c);
        s_h1[p ^ 1][(hi * 8 + r) * H_DIM + u] = (_Float16)h;
      }
    }
    __syncthreads();   // h1(t) complete & visible

    // ---------- phase 3: gates1 = h1(t) @ Wih1^T + h2_prev @ Whh1^T (K=128 fused) ----------
    #pragma unroll
    for (int g = 0; g < 4; ++g) acc[g] = vzero8f();
    {
      v16h a0 = load_a_frag(s_h1[p ^ 1], lane, 0);
      v16h a1 = load_a_frag(s_h1[p ^ 1], lane, 32);
      v16h a2 = load_a_frag(s_h2[p],     lane, 0);
      v16h a3 = load_a_frag(s_h2[p],     lane, 32);
      #pragma unroll
      for (int grp = 0; grp < 4; ++grp) {
        acc[grp] = wmma_f16(a0, wbi1[grp][0], acc[grp]);
        acc[grp] = wmma_f16(a1, wbi1[grp][1], acc[grp]);
        acc[grp] = wmma_f16(a2, wbh1[grp][0], acc[grp]);
        acc[grp] = wmma_f16(a3, wbh1[grp][1], acc[grp]);
      }
    }

    // ---------- phase 4: layer-2 elementwise ----------
    #pragma unroll
    for (int r = 0; r < 8; ++r) {
      const float iv = fast_sig(acc[0][r] + bi1);
      const float fv = fast_sig(acc[1][r] + bf1);
      const float gv = fast_tanh(acc[2][r] + bg1);
      const float ov = fast_sig(acc[3][r] + bo1);
      const float c  = fv * c2[r] + iv * gv;
      c2[r] = c;
      const float h = ov * fast_tanh(c);
      s_h2[p ^ 1][(hi * 8 + r) * H_DIM + u] = (_Float16)h;
    }
    __syncthreads();   // h2(t) complete; x(t+1) also visible
    p ^= 1;
  }

  // ---------- final FC: out = relu(h2_T @ Wfc^T + bfc) ----------
  {
    v16h a0 = load_a_frag(s_h2[p], lane, 0);
    v16h a1 = load_a_frag(s_h2[p], lane, 32);
    #pragma unroll
    for (int j = 0; j < 2; ++j) {
      const int g0 = (q * 2 + j) * 16;     // 8 out-tiles, 2 per wave
      v8f c = vzero8f();
      c = wmma_f16(a0, load_b_frag_g(Wfc, g0, lane, 0),  c);
      c = wmma_f16(a1, load_b_frag_g(Wfc, g0, lane, 32), c);
      const float bv = bfc[g0 + n];
      #pragma unroll
      for (int r = 0; r < 8; ++r) {
        float v = c[r] + bv;
        v = v > 0.f ? v : 0.f;
        out[(size_t)(b0g + hi * 8 + r) * OUT_DIM + g0 + n] = v;
      }
    }
  }
}

extern "C" void kernel_launch(void* const* d_in, const int* in_sizes, int n_in,
                              void* d_out, int out_size, void* d_ws, size_t ws_size,
                              hipStream_t stream) {
  (void)in_sizes; (void)n_in; (void)d_ws; (void)ws_size; (void)out_size;
  const float* x    = (const float*)d_in[0];
  const float* Wih0 = (const float*)d_in[1];
  const float* Whh0 = (const float*)d_in[2];
  const float* bih0 = (const float*)d_in[3];
  const float* bhh0 = (const float*)d_in[4];
  const float* Wih1 = (const float*)d_in[5];
  const float* Whh1 = (const float*)d_in[6];
  const float* bih1 = (const float*)d_in[7];
  const float* bhh1 = (const float*)d_in[8];
  const float* Wfc  = (const float*)d_in[9];
  const float* bfc  = (const float*)d_in[10];
  float* out = (float*)d_out;

  lstm_encoder_kernel<<<dim3(1024 / TILE_B), dim3(NTHREADS), 0, stream>>>(
      x, Wih0, Whh0, bih0, bhh0, Wih1, Whh1, bih1, bhh1, Wfc, bfc, out);
}
